// MultiwayFeedForwardLayer_4810363372426
// MI455X (gfx1250) — matrix operand order repaired and checked by reading, not compiled, not run on hardware
//
#include <hip/hip_runtime.h>
#include <hip/hip_bf16.h>

// ---------------------------------------------------------------------------
// MoE top-1 feed-forward:  out[t] = gelu(x[t] @ W1[e] + b1[e]) @ W2[e] + b2[e]
// Grouped-GEMM (token binning) + bf16 WMMA, double-buffered LDS pipeline,
// async global->LDS (ASYNCcnt) for the bf16 h-tile, batched loads + hw cvt.
// ---------------------------------------------------------------------------

#define BS      2
#define SEQ     2048
#define T_TOK   (BS * SEQ)          // 4096 tokens
#define DIM     1024
#define DMLP    4096
#define NEXP    8
#define TILE_M  128
#define MT      40                  // worst case tiles: 4096 + 8*127 -> 40
#define PADDED  (MT * TILE_M)       // 5120 padded rows
#define LDS_STRIDE 40               // 32 K elems + 8 pad (bf16 units)

typedef __bf16  v16bf __attribute__((ext_vector_type(16)));
typedef __bf16  v2bf  __attribute__((ext_vector_type(2)));
typedef float   v8f   __attribute__((ext_vector_type(8)));
typedef unsigned int v4u __attribute__((ext_vector_type(4)));
typedef unsigned int v8u __attribute__((ext_vector_type(8)));
typedef int     v2i   __attribute__((ext_vector_type(2)));
typedef int     v4i   __attribute__((ext_vector_type(4)));

// ---- CDNA5 async global->LDS (guarded; falls back to sync copy) -----------
#if defined(__has_builtin)
#  if __has_builtin(__builtin_amdgcn_global_load_async_to_lds_b128)
#    define HAVE_ASYNC_B128 1
#  else
#    define HAVE_ASYNC_B128 0
#  endif
#  if __has_builtin(__builtin_amdgcn_global_load_async_to_lds_b64)
#    define HAVE_ASYNC_B64 1
#  else
#    define HAVE_ASYNC_B64 0
#  endif
#else
#  define HAVE_ASYNC_B128 0
#  define HAVE_ASYNC_B64 0
#endif

typedef __attribute__((address_space(1))) v2i* g_v2i_p;   // global <2 x i32>*
typedef __attribute__((address_space(3))) v2i* l_v2i_p;   // LDS    <2 x i32>*
typedef __attribute__((address_space(1))) v4i* g_v4i_p;   // global <4 x i32>*
typedef __attribute__((address_space(3))) v4i* l_v4i_p;   // LDS    <4 x i32>*

// generic->AS1 keeps the same VA; generic->AS3 keeps low 32 bits (LDS offset)
__device__ __forceinline__ void async_cp_b128(const void* g, void* l) {
#if HAVE_ASYNC_B128
    __builtin_amdgcn_global_load_async_to_lds_b128(
        (g_v4i_p)(unsigned long long)g, (l_v4i_p)(unsigned long long)l, 0, 0);
#elif HAVE_ASYNC_B64
    __builtin_amdgcn_global_load_async_to_lds_b64(
        (g_v2i_p)(unsigned long long)g, (l_v2i_p)(unsigned long long)l, 0, 0);
    __builtin_amdgcn_global_load_async_to_lds_b64(
        (g_v2i_p)((unsigned long long)g + 8), (l_v2i_p)((unsigned long long)l + 8), 0, 0);
#else
    *(uint4*)l = *(const uint4*)g;
#endif
}

__device__ __forceinline__ void wait_async_all() {
#if HAVE_ASYNC_B128 || HAVE_ASYNC_B64
#  if __has_builtin(__builtin_amdgcn_s_wait_asynccnt)
    __builtin_amdgcn_s_wait_asynccnt(0);
#  else
    asm volatile("s_wait_asynccnt 0" ::: "memory");
#  endif
#endif
}

// ---- fp32 -> bf16 via hardware convert (v_cvt_pk_bf16_f32, RNE) -----------
__device__ __forceinline__ unsigned short f2bf(float f) {
    __bf16 b = (__bf16)f;
    return __builtin_bit_cast(unsigned short, b);
}
__device__ __forceinline__ unsigned int pack2(float a, float b) {
    v2bf v; v.x = (__bf16)a; v.y = (__bf16)b;
    return __builtin_bit_cast(unsigned int, v);
}
__device__ __forceinline__ float gelu_exact(float v) {
    return 0.5f * v * (1.0f + erff(v * 0.70710678118654752f));
}

// ---------------- binning kernels ----------------

__global__ void moe_init(int* counts, int* cursor, int* perm) {
    int i = blockIdx.x * blockDim.x + threadIdx.x;
    if (i < PADDED) perm[i] = -1;
    if (i < NEXP) { counts[i] = 0; cursor[i] = 0; }
}

__global__ void moe_count(const int* __restrict__ idx, int* counts) {
    int t = blockIdx.x * blockDim.x + threadIdx.x;
    if (t < T_TOK) atomicAdd(&counts[idx[t]], 1);
}

__global__ void moe_offsets(const int* __restrict__ counts,
                            int* aligned_off, int* tile_expert) {
    if (threadIdx.x == 0 && blockIdx.x == 0) {
        int off = 0;
        for (int e = 0; e < NEXP; ++e) {
            aligned_off[e] = off;
            int tiles = (counts[e] + TILE_M - 1) / TILE_M;
            for (int t = 0; t < tiles; ++t) tile_expert[off / TILE_M + t] = e;
            off += tiles * TILE_M;
        }
        for (int t = off / TILE_M; t < MT; ++t) tile_expert[t] = -1;
    }
}

__global__ void moe_scatter(const int* __restrict__ idx,
                            const int* __restrict__ aligned_off,
                            int* cursor, int* perm) {
    int t = blockIdx.x * blockDim.x + threadIdx.x;
    if (t < T_TOK) {
        int e = idx[t];
        int p = atomicAdd(&cursor[e], 1);
        perm[aligned_off[e] + p] = t;
    }
}

// ---------------- shared WMMA tile compute ----------------
// Tile: 128x128 per block, 8 waves, wave tile 64(M) x 32(N) = 4x2 WMMA tiles.

__device__ __forceinline__ void tile_mma(const unsigned short* sA,
                                         const unsigned short* sB,
                                         int wm, int wn, int rsel, int hsel,
                                         v8f (&acc)[4][2]) {
    v16bf afrag[4], bfrag[2];
    #pragma unroll
    for (int mi = 0; mi < 4; ++mi) {
        const unsigned short* p = &sA[(wm * 64 + mi * 16 + rsel) * LDS_STRIDE];
        v8u t;
        t.lo = *(const v4u*)(p + hsel * 8);        // K 0-7   (or 8-15)
        t.hi = *(const v4u*)(p + 16 + hsel * 8);   // K 16-23 (or 24-31)
        afrag[mi] = __builtin_bit_cast(v16bf, t);
    }
    #pragma unroll
    for (int ni = 0; ni < 2; ++ni) {
        const unsigned short* p = &sB[(wn * 32 + ni * 16 + rsel) * LDS_STRIDE + hsel * 8];
        v8u t;
        t.lo = *(const v4u*)(p);
        t.hi = *(const v4u*)(p + 16);
        bfrag[ni] = __builtin_bit_cast(v16bf, t);
    }
    #pragma unroll
    for (int mi = 0; mi < 4; ++mi)
        #pragma unroll
        for (int ni = 0; ni < 2; ++ni)
            acc[mi][ni] = __builtin_amdgcn_wmma_f32_16x16x32_bf16(
                false, afrag[mi], false, bfrag[ni],
                (short)0, acc[mi][ni], false, false);
}

// ---------------- GEMM1: h = gelu(x_perm @ W1[e] + b1[e])  (bf16 out) -------
// tokA entries are pre-clamped to >= 0 (padding rows read token 0: finite
// data, and their outputs are never stored), so the loads are unconditional.

__device__ __forceinline__ void g1_stage(const float* __restrict__ x,
                                         const float* __restrict__ W1e,
                                         const int* tokA, int tid, int n0, int k0,
                                         unsigned short* sA, unsigned short* sB) {
    // -------- phase 1: issue all global loads --------
    float4 av[4];
    float  bv[4][4];
    #pragma unroll
    for (int i = 0; i < 4; ++i) {
        int s = tid + i * 256;
        int cg = s & 7;
        av[i] = *(const float4*)(x + (size_t)tokA[i] * DIM + k0 + cg * 4);
    }
    #pragma unroll
    for (int i = 0; i < 4; ++i) {
        int s = tid + i * 256;
        int n = s & 127, kg = s >> 7;
        size_t base = (size_t)(k0 + kg * 4) * DMLP + n0 + n;
        bv[i][0] = W1e[base];
        bv[i][1] = W1e[base + DMLP];
        bv[i][2] = W1e[base + 2 * (size_t)DMLP];
        bv[i][3] = W1e[base + 3 * (size_t)DMLP];
        if (k0 + 64 < DIM)
            __builtin_prefetch(&W1e[base + 64 * (size_t)DMLP], 0, 1);
    }
    // -------- phase 2: convert + LDS store --------
    #pragma unroll
    for (int i = 0; i < 4; ++i) {
        int s = tid + i * 256;
        int row = s >> 3, cg = s & 7;
        uint2 pk; pk.x = pack2(av[i].x, av[i].y); pk.y = pack2(av[i].z, av[i].w);
        *(uint2*)(&sA[row * LDS_STRIDE + cg * 4]) = pk;
    }
    #pragma unroll
    for (int i = 0; i < 4; ++i) {
        int s = tid + i * 256;
        int n = s & 127, kg = s >> 7;
        uint2 pk; pk.x = pack2(bv[i][0], bv[i][1]); pk.y = pack2(bv[i][2], bv[i][3]);
        *(uint2*)(&sB[n * LDS_STRIDE + kg * 4]) = pk;
    }
}

__global__ __launch_bounds__(256)
void moe_gemm1(const float* __restrict__ x, const float* __restrict__ W1,
               const float* __restrict__ b1, const int* __restrict__ perm,
               const int* __restrict__ tile_expert,
               unsigned short* __restrict__ hbuf) {
    __shared__ __attribute__((aligned(16))) unsigned short sA[2][TILE_M * LDS_STRIDE];
    __shared__ __attribute__((aligned(16))) unsigned short sB[2][128 * LDS_STRIDE];

    const int tileM = blockIdx.y;
    const int e = tile_expert[tileM];
    if (e < 0) return;
    const int n0 = blockIdx.x * 128;
    const int tid  = threadIdx.x;
    const int lane = tid & 31, wave = tid >> 5;
    const int wm = wave & 1, wn = wave >> 1;
    const int rsel = lane & 15, hsel = lane >> 4;
    const float* W1e = W1 + (size_t)e * DIM * DMLP;

    int tokA[4];
    #pragma unroll
    for (int i = 0; i < 4; ++i) {
        int t = perm[tileM * TILE_M + ((tid + i * 256) >> 3)];
        tokA[i] = t < 0 ? 0 : t;                 // clamp: unconditional loads
    }

    v8f acc[4][2] = {};
    const int KT = DIM / 32;

    g1_stage(x, W1e, tokA, tid, n0, 0, sA[0], sB[0]);
    __syncthreads();
    for (int kt = 0; kt < KT; ++kt) {
        int cur = kt & 1;
        if (kt + 1 < KT)
            g1_stage(x, W1e, tokA, tid, n0, (kt + 1) * 32, sA[cur ^ 1], sB[cur ^ 1]);
        tile_mma(sA[cur], sB[cur], wm, wn, rsel, hsel, acc);
        __syncthreads();
    }

    // epilogue: bias + exact GELU, store bf16 h rows
    #pragma unroll
    for (int mi = 0; mi < 4; ++mi) {
        int m_base = tileM * TILE_M + wm * 64 + mi * 16 + hsel * 8;
        #pragma unroll
        for (int ni = 0; ni < 2; ++ni) {
            int n_g = n0 + wn * 32 + ni * 16 + rsel;
            float bias = b1[e * DMLP + n_g];
            v8f c = acc[mi][ni];
            #pragma unroll
            for (int r = 0; r < 8; ++r) {
                float v = gelu_exact(c[r] + bias);
                hbuf[(size_t)(m_base + r) * DMLP + n_g] = f2bf(v);
            }
        }
    }
}

// ---------------- GEMM2: out = h @ W2[e] + b2[e], scatter by perm -----------

__device__ __forceinline__ void g2_stage(const unsigned short* __restrict__ hbuf,
                                         const float* __restrict__ W2e,
                                         int tileM, int tid, int n0, int k0,
                                         unsigned short* sA, unsigned short* sB) {
    // A: 128 x 32 bf16 straight from hbuf — async DMA to LDS (16B per lane)
    #pragma unroll
    for (int i = 0; i < 2; ++i) {
        int s = tid + i * 256;
        int row = s >> 2, cg = s & 3;            // cg: 8-elem (16B) chunks
        async_cp_b128(hbuf + (size_t)(tileM * TILE_M + row) * DMLP + k0 + cg * 8,
                      &sA[row * LDS_STRIDE + cg * 8]);
    }
    // B: 32 x 128 of W2[e] (DMLP x DIM), stored transposed as Bt[n][k]
    float bv[4][4];
    #pragma unroll
    for (int i = 0; i < 4; ++i) {
        int s = tid + i * 256;
        int n = s & 127, kg = s >> 7;
        size_t base = (size_t)(k0 + kg * 4) * DIM + n0 + n;
        bv[i][0] = W2e[base];
        bv[i][1] = W2e[base + DIM];
        bv[i][2] = W2e[base + 2 * (size_t)DIM];
        bv[i][3] = W2e[base + 3 * (size_t)DIM];
        if (k0 + 64 < DMLP)
            __builtin_prefetch(&W2e[base + 64 * (size_t)DIM], 0, 1);
    }
    #pragma unroll
    for (int i = 0; i < 4; ++i) {
        int s = tid + i * 256;
        int n = s & 127, kg = s >> 7;
        uint2 pk; pk.x = pack2(bv[i][0], bv[i][1]); pk.y = pack2(bv[i][2], bv[i][3]);
        *(uint2*)(&sB[n * LDS_STRIDE + kg * 4]) = pk;
    }
}

__global__ __launch_bounds__(256)
void moe_gemm2(const unsigned short* __restrict__ hbuf,
               const float* __restrict__ W2, const float* __restrict__ b2,
               const int* __restrict__ perm, const int* __restrict__ tile_expert,
               float* __restrict__ out) {
    __shared__ __attribute__((aligned(16))) unsigned short sA[2][TILE_M * LDS_STRIDE];
    __shared__ __attribute__((aligned(16))) unsigned short sB[2][128 * LDS_STRIDE];

    const int tileM = blockIdx.y;
    const int e = tile_expert[tileM];
    if (e < 0) return;
    const int n0 = blockIdx.x * 128;
    const int tid  = threadIdx.x;
    const int lane = tid & 31, wave = tid >> 5;
    const int wm = wave & 1, wn = wave >> 1;
    const int rsel = lane & 15, hsel = lane >> 4;
    const float* W2e = W2 + (size_t)e * DMLP * DIM;

    v8f acc[4][2] = {};
    const int KT = DMLP / 32;

    g2_stage(hbuf, W2e, tileM, tid, n0, 0, sA[0], sB[0]);
    wait_async_all();
    __syncthreads();
    for (int kt = 0; kt < KT; ++kt) {
        int cur = kt & 1;
        if (kt + 1 < KT)
            g2_stage(hbuf, W2e, tileM, tid, n0, (kt + 1) * 32, sA[cur ^ 1], sB[cur ^ 1]);
        tile_mma(sA[cur], sB[cur], wm, wn, rsel, hsel, acc);
        wait_async_all();
        __syncthreads();
    }

    // epilogue: bias + scatter fp32 rows to out by token index
    #pragma unroll
    for (int mi = 0; mi < 4; ++mi) {
        int m_base = tileM * TILE_M + wm * 64 + mi * 16 + hsel * 8;
        #pragma unroll
        for (int ni = 0; ni < 2; ++ni) {
            int n_g = n0 + wn * 32 + ni * 16 + rsel;
            float bias = b2[e * DIM + n_g];
            v8f c = acc[mi][ni];
            #pragma unroll
            for (int r = 0; r < 8; ++r) {
                int tok = perm[m_base + r];
                if (tok >= 0)
                    out[(size_t)tok * DIM + n_g] = c[r] + bias;
            }
        }
    }
}

// ---------------- host launcher ----------------

extern "C" void kernel_launch(void* const* d_in, const int* in_sizes, int n_in,
                              void* d_out, int out_size, void* d_ws, size_t ws_size,
                              hipStream_t stream) {
    const float* x   = (const float*)d_in[0];
    const int*   idx = (const int*)d_in[1];
    const float* W1  = (const float*)d_in[2];
    const float* b1  = (const float*)d_in[3];
    const float* W2  = (const float*)d_in[4];
    const float* b2  = (const float*)d_in[5];
    float* out = (float*)d_out;

    int* counts      = (int*)d_ws;            // 8
    int* cursor      = counts + 8;            // 8
    int* aligned_off = cursor + 8;            // 8
    int* tile_expert = aligned_off + 8;       // MT (40)
    int* perm        = tile_expert + 64;      // PADDED (5120)
    unsigned short* hbuf = (unsigned short*)((char*)d_ws + 65536); // PADDED x DMLP bf16 (~40MB)

    moe_init   <<<(PADDED + 255) / 256, 256, 0, stream>>>(counts, cursor, perm);
    moe_count  <<<(T_TOK + 255) / 256, 256, 0, stream>>>(idx, counts);
    moe_offsets<<<1, 32, 0, stream>>>(counts, aligned_off, tile_expert);
    moe_scatter<<<(T_TOK + 255) / 256, 256, 0, stream>>>(idx, aligned_off, cursor, perm);

    moe_gemm1<<<dim3(DMLP / 128, MT), 256, 0, stream>>>(x, W1, b1, perm, tile_expert, hbuf);
    moe_gemm2<<<dim3(DIM  / 128, MT), 256, 0, stream>>>(hbuf, W2, b2, perm, tile_expert, out);
}
